// AttentionHead_54279796687221
// MI455X (gfx1250) — matrix operand order repaired and checked by reading, not compiled, not run on hardware
//
#include <hip/hip_runtime.h>
#include <math.h>

typedef __attribute__((ext_vector_type(16))) __bf16 v16bf;
typedef __attribute__((ext_vector_type(8)))  __bf16 v8bf;
typedef __attribute__((ext_vector_type(8)))  float  v8f;

#define EMBED 768
#define DK    64
#define BATCH 4
#define SEQ   4096
#define ROWS  (BATCH * SEQ)   // 16384

#define USE_ASYNC_LDS 1       // gfx1250 async global->LDS path (assembled OK in round 2)

static __device__ __forceinline__ v8f wmma_bf16(v16bf a, v16bf b, v8f c) {
    // D = A(16x32 bf16) * B(32x16 bf16) + C(16x16 f32)
    return __builtin_amdgcn_wmma_f32_16x16x32_bf16(
        /*neg_a=*/false, a, /*neg_b=*/false, b,
        /*c_mod=*/(short)0, c, /*reuse_a=*/false, /*reuse_b=*/false);
}

// Load 16 contiguous bf16 as v16bf via two 16B loads (LDS pitch gives 16B align).
static __device__ __forceinline__ v16bf load16(const __bf16* p) {
    v8bf lo = *(const v8bf*)p;
    v8bf hv = *(const v8bf*)(p + 8);
    v16bf r;
    #pragma unroll
    for (int j = 0; j < 8; ++j) { r[j] = lo[j]; r[j + 8] = hv[j]; }
    return r;
}

static __device__ __forceinline__ void async_wait0() {
#if __has_builtin(__builtin_amdgcn_s_wait_asynccnt)
    __builtin_amdgcn_s_wait_asynccnt(0);
#else
    asm volatile("s_wait_asynccnt 0x0" ::: "memory");
#endif
}
static __device__ __forceinline__ void async_wait8() {
#if __has_builtin(__builtin_amdgcn_s_wait_asynccnt)
    __builtin_amdgcn_s_wait_asynccnt(8);
#else
    asm volatile("s_wait_asynccnt 0x8" ::: "memory");
#endif
}

// ---------------------------------------------------------------------------
// Kernel 0: W [768,64] fp32 -> Wt [64,768] bf16 (transpose + convert).
// ---------------------------------------------------------------------------
__global__ void wt_convert_kernel(const float* __restrict__ W, __bf16* __restrict__ Wt) {
    int idx = blockIdx.x * 256 + threadIdx.x;      // exactly EMBED*DK threads
    if (idx >= EMBED * DK) return;
    int e = idx / DK;
    int d = idx % DK;
    Wt[d * EMBED + e] = (__bf16)W[idx];
}

// ---------------------------------------------------------------------------
// Kernel 1: fused QKV projection. One wave -> 16 rows x 64 dims of Q,K,V.
// Q gets the 1/sqrt(64) score scale folded in. V stored transposed [B,64,S].
// ---------------------------------------------------------------------------
__global__ __launch_bounds__(128)
void qkv_proj_kernel(const float* __restrict__ x,
                     const __bf16* __restrict__ Wqt,
                     const __bf16* __restrict__ Wkt,
                     const __bf16* __restrict__ Wvt,
                     const float* __restrict__ bq,
                     const float* __restrict__ bk,
                     const float* __restrict__ bv,
                     __bf16* __restrict__ Q,
                     __bf16* __restrict__ K,
                     __bf16* __restrict__ Vt) {
    const int wave = threadIdx.x >> 5;
    const int lane = threadIdx.x & 31;
    const int tile = blockIdx.x * 4 + wave;        // 1024 tiles of 16 rows
    const int r0   = tile * 16;
    const int m    = lane & 15;
    const int hi   = lane >> 4;

    v8f cq[4], ck[4], cv[4];
    for (int t = 0; t < 4; ++t) { cq[t] = v8f{}; ck[t] = v8f{}; cv[t] = v8f{}; }

    const float* xrow = x + (size_t)(r0 + m) * EMBED;

    #pragma unroll 2
    for (int kb = 0; kb < EMBED; kb += 32) {
        // A fragment: 16x32 bf16 slice of x (convert fp32 -> bf16 in regs)
        v16bf a;
        const float* p0 = xrow + kb + hi * 16;
        const float* p1 = xrow + kb + hi * 16 + 8;
        #pragma unroll
        for (int j = 0; j < 8; ++j) { a[j] = (__bf16)p0[j]; a[j + 8] = (__bf16)p1[j]; }

        #pragma unroll
        for (int t = 0; t < 4; ++t) {
            const int d = t * 16 + m;
            const size_t woff = (size_t)d * EMBED + kb + hi * 16;
            v16bf bqf = *(const v16bf*)(Wqt + woff);
            v16bf bkf = *(const v16bf*)(Wkt + woff);
            v16bf bvf = *(const v16bf*)(Wvt + woff);
            cq[t] = wmma_bf16(a, bqf, cq[t]);
            ck[t] = wmma_bf16(a, bkf, ck[t]);
            cv[t] = wmma_bf16(a, bvf, cv[t]);
        }
    }

    // Epilogue: bias, fold 0.125 into Q, convert to bf16, store.
    #pragma unroll
    for (int t = 0; t < 4; ++t) {
        const int d = t * 16 + m;
        const float bqv = bq[d], bkv = bk[d], bvv = bv[d];
        #pragma unroll
        for (int r = 0; r < 8; ++r) {
            const int grow = r0 + r + 8 * hi;            // global row (D layout)
            Q[(size_t)grow * DK + d] = (__bf16)((cq[t][r] + bqv) * 0.125f);
            K[(size_t)grow * DK + d] = (__bf16)(ck[t][r] + bkv);
            const int b = grow >> 12;                    // / SEQ
            const int s = grow & (SEQ - 1);
            Vt[((size_t)(b * DK + d)) * SEQ + s] = (__bf16)(cv[t][r] + bvv);
        }
    }
}

// ---------------------------------------------------------------------------
// Kernel 2: flash attention, 64-key chunks, double-buffered block-cooperative
// K/V staging via async global->LDS (ASYNCcnt in-order completion lets us
// wait <=8 to cover the current chunk while the next chunk is in flight).
// ---------------------------------------------------------------------------
#define KC   64            // keys per chunk
#define LP   72            // padded LDS pitch (bf16): 144B rows, conflict-free

typedef __bf16 KsBuf[KC][LP];
typedef __bf16 VsBuf[DK][LP];

static __device__ __forceinline__ void stage_chunk(const __bf16* __restrict__ Kb,
                                                   const __bf16* __restrict__ Vb,
                                                   int kb, __bf16 (*ks)[LP], __bf16 (*vs)[LP],
                                                   int tid) {
    #pragma unroll
    for (int i = 0; i < 4; ++i) {
        const int idx = tid + i * 128;             // 0..511
        const int row = idx >> 3;                  // 0..63
        const int seg = idx & 7;                   // 8 bf16 (16B) segments
        const __bf16* gk = Kb + (size_t)(kb + row) * DK + seg * 8;
        const __bf16* gv = Vb + (size_t)row * SEQ + kb + seg * 8;
#if USE_ASYNC_LDS
        unsigned lk = (unsigned)(uintptr_t)&ks[row][seg * 8];
        unsigned lv = (unsigned)(uintptr_t)&vs[row][seg * 8];
        asm volatile("global_load_async_to_lds_b128 %0, %1, off"
                     :: "v"(lk), "v"(gk) : "memory");
        asm volatile("global_load_async_to_lds_b128 %0, %1, off"
                     :: "v"(lv), "v"(gv) : "memory");
#else
        *(v8bf*)&ks[row][seg * 8] = *(const v8bf*)gk;
        *(v8bf*)&vs[row][seg * 8] = *(const v8bf*)gv;
#endif
    }
}

__global__ __launch_bounds__(128)
void attn_kernel(const __bf16* __restrict__ Q,
                 const __bf16* __restrict__ K,
                 const __bf16* __restrict__ Vt,
                 float* __restrict__ out) {
    __shared__ __align__(16) __bf16 Ks[2][KC][LP];     // double-buffered K chunk
    __shared__ __align__(16) __bf16 Vs[2][DK][LP];     // double-buffered V^T chunk
    __shared__ __align__(16) __bf16 Ps[4][16][LP];     // per-wave P tiles

    const int tid  = threadIdx.x;
    const int wave = tid >> 5;
    const int lane = tid & 31;
    const int q0   = blockIdx.x * 64 + wave * 16;      // global query row of tile
    const int b    = q0 >> 12;                         // batch (block-uniform)
    const int m    = lane & 15;
    const int hi   = lane >> 4;

    // --- Load Q A-fragments (d=64 -> two 16x32 fragments) ---
    const __bf16* qrow = Q + (size_t)(q0 + m) * DK;
    v16bf aq0, aq1;
    {
        v8bf l0 = *(const v8bf*)(qrow + hi * 16);
        v8bf h0 = *(const v8bf*)(qrow + 8 + hi * 16);
        v8bf l1 = *(const v8bf*)(qrow + 32 + hi * 16);
        v8bf h1 = *(const v8bf*)(qrow + 40 + hi * 16);
        #pragma unroll
        for (int j = 0; j < 8; ++j) {
            aq0[j] = l0[j]; aq0[j + 8] = h0[j];
            aq1[j] = l1[j]; aq1[j + 8] = h1[j];
        }
    }

    float mrow[8], lrow[8];
    v8f acc[4];
    #pragma unroll
    for (int r = 0; r < 8; ++r) { mrow[r] = -3.0e38f; lrow[r] = 0.0f; }
    #pragma unroll
    for (int t = 0; t < 4; ++t) acc[t] = v8f{};

    const __bf16* Kb = K  + (size_t)b * SEQ * DK;
    const __bf16* Vb = Vt + (size_t)b * DK * SEQ;

    // Prologue: stage chunk 0 into buffer 0.
    stage_chunk(Kb, Vb, 0, Ks[0], Vs[0], tid);

    int p = 0;
    for (int kb = 0; kb < SEQ; kb += KC, p ^= 1) {
        // Issue next chunk into the other buffer (safe: its prior readers
        // passed the trailing barrier of the previous iteration), then wait
        // only for the current chunk's 8 in-order async loads.
        if (kb + KC < SEQ) {
            stage_chunk(Kb, Vb, kb + KC, Ks[p ^ 1], Vs[p ^ 1], tid);
#if USE_ASYNC_LDS
            async_wait8();
#endif
        } else {
#if USE_ASYNC_LDS
            async_wait0();
#endif
        }
        __syncthreads();

        // ---- Scores: 4 D-tiles of 16x16, each = Q(16x64) . K^T ----
        v8f s[4];
        #pragma unroll
        for (int j = 0; j < 4; ++j) {
            const __bf16* kr = &Ks[p][16 * j + m][0];
            s[j] = v8f{};
            s[j] = wmma_bf16(aq0, load16(kr + hi * 16),      s[j]);
            s[j] = wmma_bf16(aq1, load16(kr + 32 + hi * 16), s[j]);
        }

        // ---- Online softmax in D layout: row = r + 8*hi, col = lane%16 ----
        #pragma unroll
        for (int r = 0; r < 8; ++r) {
            float tmax = fmaxf(fmaxf(s[0][r], s[1][r]), fmaxf(s[2][r], s[3][r]));
            #pragma unroll
            for (int off = 8; off >= 1; off >>= 1)
                tmax = fmaxf(tmax, __shfl_xor(tmax, off, 16));
            const float nm   = fmaxf(mrow[r], tmax);
            const float corr = __expf(mrow[r] - nm);
            float p0 = __expf(s[0][r] - nm);
            float p1 = __expf(s[1][r] - nm);
            float p2 = __expf(s[2][r] - nm);
            float p3 = __expf(s[3][r] - nm);
            float rs = (p0 + p1) + (p2 + p3);
            #pragma unroll
            for (int off = 8; off >= 1; off >>= 1)
                rs += __shfl_xor(rs, off, 16);
            lrow[r] = lrow[r] * corr + rs;
            mrow[r] = nm;
            #pragma unroll
            for (int t = 0; t < 4; ++t) acc[t][r] *= corr;
            __bf16* pr = &Ps[wave][r + 8 * hi][0];
            pr[m]      = (__bf16)p0;
            pr[m + 16] = (__bf16)p1;
            pr[m + 32] = (__bf16)p2;
            pr[m + 48] = (__bf16)p3;
        }
        // Same-wave DS ops are in-order; per-wave P region needs no barrier.

        // ---- acc += P(16x64) . V(64x64): 2 K-subchunks x 4 N-tiles ----
        #pragma unroll
        for (int c = 0; c < 2; ++c) {
            const __bf16* pr = &Ps[wave][m][32 * c];
            v16bf ap;
            {
                v8bf lo = *(const v8bf*)(pr + hi * 16);
                v8bf hv = *(const v8bf*)(pr + hi * 16 + 8);
                #pragma unroll
                for (int j = 0; j < 8; ++j) { ap[j] = lo[j]; ap[j + 8] = hv[j]; }
            }
            #pragma unroll
            for (int t = 0; t < 4; ++t)
                acc[t] = wmma_bf16(ap, load16(&Vs[p][16 * t + m][32 * c + 16 * hi]), acc[t]);
        }
        __syncthreads();   // all waves done with Ks[p]/Vs[p] before it is restaged
    }

    // ---- Epilogue: normalize by softmax denominator, store fp32 ----
    #pragma unroll
    for (int r = 0; r < 8; ++r) {
        const float inv  = 1.0f / lrow[r];
        const int   grow = q0 + r + 8 * hi;
        #pragma unroll
        for (int t = 0; t < 4; ++t)
            out[(size_t)grow * DK + t * 16 + m] = acc[t][r] * inv;
    }
}

// ---------------------------------------------------------------------------
// Host launcher
// ---------------------------------------------------------------------------
extern "C" void kernel_launch(void* const* d_in, const int* in_sizes, int n_in,
                              void* d_out, int out_size, void* d_ws, size_t ws_size,
                              hipStream_t stream) {
    const float* x  = (const float*)d_in[0];
    const float* Wq = (const float*)d_in[1];
    const float* bq = (const float*)d_in[2];
    const float* Wk = (const float*)d_in[3];
    const float* bk = (const float*)d_in[4];
    const float* Wv = (const float*)d_in[5];
    const float* bv = (const float*)d_in[6];
    float* out = (float*)d_out;

    // Workspace layout (bf16 units): Wqt|Wkt|Wvt|Q|K|Vt  (~6.6 MB total)
    __bf16* ws  = (__bf16*)d_ws;
    __bf16* Wqt = ws;
    __bf16* Wkt = Wqt + (size_t)EMBED * DK;
    __bf16* Wvt = Wkt + (size_t)EMBED * DK;
    __bf16* Qb  = Wvt + (size_t)EMBED * DK;
    __bf16* Kb  = Qb  + (size_t)ROWS * DK;
    __bf16* Vt  = Kb  + (size_t)ROWS * DK;

    wt_convert_kernel<<<(EMBED * DK) / 256, 256, 0, stream>>>(Wq, Wqt);
    wt_convert_kernel<<<(EMBED * DK) / 256, 256, 0, stream>>>(Wk, Wkt);
    wt_convert_kernel<<<(EMBED * DK) / 256, 256, 0, stream>>>(Wv, Wvt);

    // 1024 row-tiles of 16, 4 waves/block
    qkv_proj_kernel<<<256, 128, 0, stream>>>(x, Wqt, Wkt, Wvt, bq, bk, bv, Qb, Kb, Vt);

    // 256 blocks x 64 queries (4 waves of 16)
    attn_kernel<<<256, 128, 0, stream>>>(Qb, Kb, Vt, out);
}
